// HeterogeneityLSTMModel_34737695490712
// MI455X (gfx1250) — matrix operand order repaired and checked by reading, not compiled, not run on hardware
//
#include <hip/hip_runtime.h>
#include <hip/hip_bf16.h>

typedef __bf16 bf16_t;
typedef __attribute__((ext_vector_type(16))) __bf16 v16bf;
typedef __attribute__((ext_vector_type(8)))  __bf16 v8bf;
typedef __attribute__((ext_vector_type(8)))  float  v8f;

#define BATCH 256
#define TSTEPS 256
#define DIN 128
#define HID 512
#define FOURH 2048
#define BH (BATCH * HID)
#define NBLK 64   // persistent grid size: 8 M-blocks x 8 J-blocks (must be co-resident)

// ---------------- workspace layout (bytes) ----------------
#define OFF_W0   0u              // W0cat bf16 [2048][640]
#define OFF_W1   2621440u        // W1cat bf16 [2048][1024]
#define OFF_H0   6815744u        // H0bf  bf16 [2][256*512]
#define OFF_H1   7340032u        // H1bf  bf16 [2][256*512]
#define OFF_HF   7864320u        // hl1 f32 [256*512]
#define OFF_B0   8388608u        // bias0 f32 [2048]
#define OFF_B1   8396800u        // bias1 f32 [2048]
#define OFF_CNT  8404992u        // grid-barrier counter (64 u32, only [0] used)
// total ~8.4 MB

__device__ __forceinline__ float sigmoidf_(float x) {
    return 1.0f / (1.0f + __expf(-x));
}

__device__ __forceinline__ v8bf cvt8_f32_bf16(v8f f) {
    v8bf o;
#pragma unroll
    for (int i = 0; i < 8; ++i) o[i] = (__bf16)f[i];
    return o;
}

// ------------------------------------------------------------------
// Prologue: bf16 concatenated weights, combined biases, zero h state + barrier
// ------------------------------------------------------------------
#define SEG0 (FOURH * 640)
#define SEG1 (FOURH * 1024)
#define SEG2 FOURH
#define SEG3 FOURH
#define SEG4 (2 * BH)
#define SEG5 (2 * BH)
#define SEG6 64
#define PREP_N (SEG0 + SEG1 + SEG2 + SEG3 + SEG4 + SEG5 + SEG6)

__global__ __launch_bounds__(256) void prep_kernel(
    const float* __restrict__ Wih0, const float* __restrict__ Whh0,
    const float* __restrict__ bih0, const float* __restrict__ bhh0,
    const float* __restrict__ Wih1, const float* __restrict__ Whh1,
    const float* __restrict__ bih1, const float* __restrict__ bhh1,
    bf16_t* __restrict__ W0cat, bf16_t* __restrict__ W1cat,
    float* __restrict__ bias0, float* __restrict__ bias1,
    bf16_t* __restrict__ H0bf, bf16_t* __restrict__ H1bf,
    unsigned* __restrict__ cnt)
{
    long idx = (long)blockIdx.x * blockDim.x + threadIdx.x;
    if (idx < SEG0) {
        int n = (int)(idx / 640), k = (int)(idx % 640);
        float v = (k < DIN) ? Wih0[n * DIN + k] : Whh0[n * HID + (k - DIN)];
        W0cat[idx] = (bf16_t)v;
        return;
    }
    idx -= SEG0;
    if (idx < SEG1) {
        int n = (int)(idx / 1024), k = (int)(idx % 1024);
        float v = (k < HID) ? Wih1[n * HID + k] : Whh1[n * HID + (k - HID)];
        W1cat[idx] = (bf16_t)v;
        return;
    }
    idx -= SEG1;
    if (idx < SEG2) { bias0[idx] = bih0[idx] + bhh0[idx]; return; }
    idx -= SEG2;
    if (idx < SEG3) { bias1[idx] = bih1[idx] + bhh1[idx]; return; }
    idx -= SEG3;
    if (idx < SEG4) { H0bf[idx] = (bf16_t)0.0f; return; }
    idx -= SEG4;
    if (idx < SEG5) { H1bf[idx] = (bf16_t)0.0f; return; }
    idx -= SEG5;
    if (idx < SEG6) { cnt[idx] = 0u; return; }
}

// ------------------------------------------------------------------
// Device-wide barrier (monotonic counter; NBLK co-resident blocks)
// ------------------------------------------------------------------
__device__ __forceinline__ void grid_sync(unsigned* cnt, unsigned target) {
    __threadfence();              // release my stores to device scope
    __syncthreads();
    if (threadIdx.x == 0) {
        __hip_atomic_fetch_add(cnt, 1u, __ATOMIC_ACQ_REL, __HIP_MEMORY_SCOPE_AGENT);
        while (__hip_atomic_load(cnt, __ATOMIC_ACQUIRE, __HIP_MEMORY_SCOPE_AGENT) < target) {
            __builtin_amdgcn_s_sleep(2);
        }
    }
    __syncthreads();
    __threadfence();              // acquire: invalidate before reading peers' h
}

// ------------------------------------------------------------------
// One LSTM cell phase for this wave's fixed 16(M)x16(J) tile.
// LAYER=0: K = [x_t(128, f32->bf16) | Hself(512)]
// LAYER=1: K = [Hlow(512)           | Hself(512)]
// ------------------------------------------------------------------
template<int LAYER>
__device__ __forceinline__ void cell_phase(
    int t, const float* __restrict__ x,
    const bf16_t* __restrict__ Wcat, const float bi[4],
    const bf16_t* __restrict__ Hself, const bf16_t* __restrict__ Hlow,
    float (&creg)[8],
    bf16_t* __restrict__ Hout, float* __restrict__ hf_out,
    int m0, int j0, int lane)
{
    const int half = lane >> 4;
    const int r    = lane & 15;
    const int sW   = LAYER ? 1024 : 640;

    v8f acc[4];
#pragma unroll
    for (int g = 0; g < 4; ++g)
#pragma unroll
        for (int e = 0; e < 8; ++e) acc[g][e] = 0.0f;

    // per-gate weight base pointers (hoisted; advance by k inside loops)
    const bf16_t* bp0 = Wcat + (size_t)(0 * HID + j0 + r) * sW + half * 16;
    const bf16_t* bp1 = Wcat + (size_t)(1 * HID + j0 + r) * sW + half * 16;
    const bf16_t* bp2 = Wcat + (size_t)(2 * HID + j0 + r) * sW + half * 16;
    const bf16_t* bp3 = Wcat + (size_t)(3 * HID + j0 + r) * sW + half * 16;

    union AF { v16bf v; v8bf h[2]; } a;

    if (LAYER == 0) {
        // --- segment 1: x_t, K=128 (4 k-blocks), f32 -> bf16 on the fly ---
        const float* xs = x + (size_t)(m0 + r) * TSTEPS * DIN + (size_t)t * DIN + half * 16;
#pragma unroll
        for (int kb = 0; kb < 4; ++kb) {
            const int k0 = kb * 32;
            v8f f0 = *(const v8f*)(xs + k0);
            v8f f1 = *(const v8f*)(xs + k0 + 8);
            v16bf b0 = *(const v16bf*)(bp0 + k0);
            v16bf b1 = *(const v16bf*)(bp1 + k0);
            v16bf b2 = *(const v16bf*)(bp2 + k0);
            v16bf b3 = *(const v16bf*)(bp3 + k0);
            a.h[0] = cvt8_f32_bf16(f0);
            a.h[1] = cvt8_f32_bf16(f1);
            acc[0] = __builtin_amdgcn_wmma_f32_16x16x32_bf16(false, a.v, false, b0, (short)0, acc[0], false, false);
            acc[1] = __builtin_amdgcn_wmma_f32_16x16x32_bf16(false, a.v, false, b1, (short)0, acc[1], false, false);
            acc[2] = __builtin_amdgcn_wmma_f32_16x16x32_bf16(false, a.v, false, b2, (short)0, acc[2], false, false);
            acc[3] = __builtin_amdgcn_wmma_f32_16x16x32_bf16(false, a.v, false, b3, (short)0, acc[3], false, false);
        }
        // --- segment 2: h_prev (this layer), K=512 ---
        const bf16_t* hs = Hself + (size_t)(m0 + r) * HID + half * 16;
#pragma unroll 4
        for (int kb = 0; kb < 16; ++kb) {
            const int k0 = kb * 32;
            a.h[0] = *(const v8bf*)(hs + k0);
            a.h[1] = *(const v8bf*)(hs + k0 + 8);
            v16bf b0 = *(const v16bf*)(bp0 + DIN + k0);
            v16bf b1 = *(const v16bf*)(bp1 + DIN + k0);
            v16bf b2 = *(const v16bf*)(bp2 + DIN + k0);
            v16bf b3 = *(const v16bf*)(bp3 + DIN + k0);
            acc[0] = __builtin_amdgcn_wmma_f32_16x16x32_bf16(false, a.v, false, b0, (short)0, acc[0], false, false);
            acc[1] = __builtin_amdgcn_wmma_f32_16x16x32_bf16(false, a.v, false, b1, (short)0, acc[1], false, false);
            acc[2] = __builtin_amdgcn_wmma_f32_16x16x32_bf16(false, a.v, false, b2, (short)0, acc[2], false, false);
            acc[3] = __builtin_amdgcn_wmma_f32_16x16x32_bf16(false, a.v, false, b3, (short)0, acc[3], false, false);
        }
    } else {
        // --- segment 1: h from layer0 (just produced), K=512 ---
        const bf16_t* hsL = Hlow + (size_t)(m0 + r) * HID + half * 16;
#pragma unroll 4
        for (int kb = 0; kb < 16; ++kb) {
            const int k0 = kb * 32;
            a.h[0] = *(const v8bf*)(hsL + k0);
            a.h[1] = *(const v8bf*)(hsL + k0 + 8);
            v16bf b0 = *(const v16bf*)(bp0 + k0);
            v16bf b1 = *(const v16bf*)(bp1 + k0);
            v16bf b2 = *(const v16bf*)(bp2 + k0);
            v16bf b3 = *(const v16bf*)(bp3 + k0);
            acc[0] = __builtin_amdgcn_wmma_f32_16x16x32_bf16(false, a.v, false, b0, (short)0, acc[0], false, false);
            acc[1] = __builtin_amdgcn_wmma_f32_16x16x32_bf16(false, a.v, false, b1, (short)0, acc[1], false, false);
            acc[2] = __builtin_amdgcn_wmma_f32_16x16x32_bf16(false, a.v, false, b2, (short)0, acc[2], false, false);
            acc[3] = __builtin_amdgcn_wmma_f32_16x16x32_bf16(false, a.v, false, b3, (short)0, acc[3], false, false);
        }
        // --- segment 2: h_prev (this layer), K=512 ---
        const bf16_t* hs = Hself + (size_t)(m0 + r) * HID + half * 16;
#pragma unroll 4
        for (int kb = 0; kb < 16; ++kb) {
            const int k0 = kb * 32;
            a.h[0] = *(const v8bf*)(hs + k0);
            a.h[1] = *(const v8bf*)(hs + k0 + 8);
            v16bf b0 = *(const v16bf*)(bp0 + HID + k0);
            v16bf b1 = *(const v16bf*)(bp1 + HID + k0);
            v16bf b2 = *(const v16bf*)(bp2 + HID + k0);
            v16bf b3 = *(const v16bf*)(bp3 + HID + k0);
            acc[0] = __builtin_amdgcn_wmma_f32_16x16x32_bf16(false, a.v, false, b0, (short)0, acc[0], false, false);
            acc[1] = __builtin_amdgcn_wmma_f32_16x16x32_bf16(false, a.v, false, b1, (short)0, acc[1], false, false);
            acc[2] = __builtin_amdgcn_wmma_f32_16x16x32_bf16(false, a.v, false, b2, (short)0, acc[2], false, false);
            acc[3] = __builtin_amdgcn_wmma_f32_16x16x32_bf16(false, a.v, false, b3, (short)0, acc[3], false, false);
        }
    }

    // ---- epilogue: activations + register-resident c update ----
    const int j = j0 + r;
#pragma unroll
    for (int e = 0; e < 8; ++e) {
        const int m = m0 + half * 8 + e;
        const size_t o = (size_t)m * HID + j;
        float iv = sigmoidf_(acc[0][e] + bi[0]);
        float fv = sigmoidf_(acc[1][e] + bi[1]);
        float gv = tanhf(acc[2][e] + bi[2]);
        float ov = sigmoidf_(acc[3][e] + bi[3]);
        float cn = fv * creg[e] + iv * gv;
        creg[e] = cn;
        float hn = ov * tanhf(cn);
        Hout[o] = (bf16_t)hn;
        if (LAYER == 1 && t == TSTEPS - 1) hf_out[o] = hn;
    }
}

// ------------------------------------------------------------------
// Persistent 2-layer LSTM: 64 blocks x 256 threads; 2 grid barriers/step.
// Each wave owns one 16x16 (m, j) tile for the entire sequence; c stays
// in VGPRs for both layers.
// ------------------------------------------------------------------
__global__ __launch_bounds__(256) void lstm_persistent_kernel(
    const float* __restrict__ x,
    const bf16_t* __restrict__ W0cat, const bf16_t* __restrict__ W1cat,
    const float* __restrict__ bias0, const float* __restrict__ bias1,
    bf16_t* __restrict__ H0bf, bf16_t* __restrict__ H1bf,
    float* __restrict__ hf_out, unsigned* __restrict__ cnt)
{
    const int lane = threadIdx.x & 31;
    const int wave = threadIdx.x >> 5;
    const int blkM = blockIdx.x & 7;
    const int blkJ = blockIdx.x >> 3;
    const int m0 = blkM * 32 + (wave & 1) * 16;
    const int j0 = blkJ * 64 + (wave >> 1) * 16;
    const int j  = j0 + (lane & 15);

    float bi0[4], bi1[4];
#pragma unroll
    for (int g = 0; g < 4; ++g) {
        bi0[g] = bias0[g * HID + j];
        bi1[g] = bias1[g * HID + j];
    }

    float c0[8], c1[8];
#pragma unroll
    for (int e = 0; e < 8; ++e) { c0[e] = 0.0f; c1[e] = 0.0f; }

    unsigned target = 0;
    for (int t = 0; t < TSTEPS; ++t) {
        const int p = t & 1, q = p ^ 1;
        cell_phase<0>(t, x, W0cat, bi0,
                      H0bf + (size_t)p * BH, nullptr, c0,
                      H0bf + (size_t)q * BH, nullptr, m0, j0, lane);
        target += NBLK;
        grid_sync(cnt, target);
        cell_phase<1>(t, x, W1cat, bi1,
                      H1bf + (size_t)p * BH, H0bf + (size_t)q * BH, c1,
                      H1bf + (size_t)q * BH, hf_out, m0, j0, lane);
        target += NBLK;
        grid_sync(cnt, target);
    }
}

// ------------------------------------------------------------------
// Final FC: out[b][cls] = [h_l1 | emb[sector]] . fc_w[cls] + fc_b[cls]
// ------------------------------------------------------------------
__global__ __launch_bounds__(256) void fc_kernel(
    const float* __restrict__ hf, const int* __restrict__ sectors,
    const float* __restrict__ emb, const float* __restrict__ fc_w,
    const float* __restrict__ fc_b, float* __restrict__ out)
{
    int idx = blockIdx.x * blockDim.x + threadIdx.x;
    if (idx >= BATCH * 10) return;
    int b = idx / 10, cls = idx % 10;
    const float* w = fc_w + cls * (HID + 16);
    float s = fc_b[cls];
    const float* hb = hf + (size_t)b * HID;
    for (int k = 0; k < HID; ++k) s += hb[k] * w[k];
    int sec = sectors[b];
    for (int k = 0; k < 16; ++k) s += emb[sec * 16 + k] * w[HID + k];
    out[idx] = s;
}

extern "C" void kernel_launch(void* const* d_in, const int* in_sizes, int n_in,
                              void* d_out, int out_size, void* d_ws, size_t ws_size,
                              hipStream_t stream) {
    const float* x     = (const float*)d_in[0];
    const int* sectors = (const int*)d_in[1];
    const float* Wih0  = (const float*)d_in[2];
    const float* Whh0  = (const float*)d_in[3];
    const float* bih0  = (const float*)d_in[4];
    const float* bhh0  = (const float*)d_in[5];
    const float* Wih1  = (const float*)d_in[6];
    const float* Whh1  = (const float*)d_in[7];
    const float* bih1  = (const float*)d_in[8];
    const float* bhh1  = (const float*)d_in[9];
    const float* emb   = (const float*)d_in[10];
    const float* fc_w  = (const float*)d_in[11];
    const float* fc_b  = (const float*)d_in[12];

    char* ws = (char*)d_ws;
    bf16_t* W0cat   = (bf16_t*)(ws + OFF_W0);
    bf16_t* W1cat   = (bf16_t*)(ws + OFF_W1);
    bf16_t* H0bf    = (bf16_t*)(ws + OFF_H0);
    bf16_t* H1bf    = (bf16_t*)(ws + OFF_H1);
    float*  hl1f    = (float*)(ws + OFF_HF);
    float*  bias0   = (float*)(ws + OFF_B0);
    float*  bias1   = (float*)(ws + OFF_B1);
    unsigned* cnt   = (unsigned*)(ws + OFF_CNT);

    int prep_blocks = (PREP_N + 255) / 256;
    prep_kernel<<<prep_blocks, 256, 0, stream>>>(
        Wih0, Whh0, bih0, bhh0, Wih1, Whh1, bih1, bhh1,
        W0cat, W1cat, bias0, bias1, H0bf, H1bf, cnt);

    lstm_persistent_kernel<<<NBLK, 256, 0, stream>>>(
        x, W0cat, W1cat, bias0, bias1, H0bf, H1bf, hl1f, cnt);

    fc_kernel<<<10, 256, 0, stream>>>(hl1f, sectors, emb, fc_w, fc_b, (float*)d_out);
}
